// Decoder_10144712753163
// MI455X (gfx1250) — compile-verified, matrix operand order checked by reading, not run on hardware
//
#include <hip/hip_runtime.h>

typedef __attribute__((ext_vector_type(16))) __bf16 v16bf;
typedef __attribute__((ext_vector_type(8)))  float  v8f;

#define VOCAB  32000
#define EMBED  512
#define HIDDEN 1024
#define MAXLEN 64
#define NB     64          // batch
#define G3     (3 * HIDDEN)

#define KP     128         // K panel staged in LDS
#define APITCH (KP + 8)    // bf16 elements; 272 B row pitch -> minimal-conflict ds reads

__device__ __forceinline__ unsigned short f2bf(float f) {
  union { float f; unsigned u; } v; v.f = f;
  unsigned r = v.u + 0x7fffu + ((v.u >> 16) & 1u);   // round-to-nearest-even
  return (unsigned short)(r >> 16);
}

#define WMMA_BF16(a, b, c) \
  __builtin_amdgcn_wmma_f32_16x16x32_bf16(false, (a), false, (b), (short)0, (c), false, false)

// CDNA5 async memory->LDS copy (ASYNCcnt-tracked), ISA 15.18.3 op 98.
// LDS address comes from the vdst VGPR; low 32 bits of a generic LDS pointer are the
// wave-relative LDS byte offset, so (unsigned)(uintptr_t)&shared[i] is valid.
__device__ __forceinline__ void async_ld_lds_b128(unsigned lds_addr, const void* gptr) {
  asm volatile("global_load_async_to_lds_b128 %0, %1, off"
               :: "v"(lds_addr), "v"(gptr) : "memory");
}
__device__ __forceinline__ void wait_async0() {
  asm volatile("s_wait_asynccnt 0x0" ::: "memory");
}

// ---------------------------------------------------------------- converters
__global__ void cvt_bf16_kernel(const float* __restrict__ src,
                                unsigned short* __restrict__ dst, int n) {
  int i = blockIdx.x * blockDim.x + threadIdx.x;
  if (i < n) dst[i] = f2bf(src[i]);
}

// ------------------------------------------------------------------- init
__global__ void init_kernel(const float* __restrict__ enc,
                            float* __restrict__ out,
                            float* __restrict__ h,
                            unsigned short* __restrict__ hb,
                            int* __restrict__ tok) {
  int i = blockIdx.x * blockDim.x + threadIdx.x;
  if (i < NB * VOCAB) {
    int b = i / VOCAB, v = i % VOCAB;
    out[(size_t)b * (MAXLEN + 1) * VOCAB + (size_t)MAXLEN * VOCAB + v] = 0.0f;
  }
  if (i < NB * HIDDEN) { float x = enc[i]; h[i] = x; hb[i] = f2bf(x); }
  if (i < NB) tok[i] = 1; // SOS
}

// ------------------------------------------------------------------ embed
__global__ void embed_kernel(const float* __restrict__ emb,
                             const int* __restrict__ tok,
                             unsigned short* __restrict__ xb) {
  int i = blockIdx.x * blockDim.x + threadIdx.x; // NB*EMBED threads
  int b = i / EMBED, e = i % EMBED;
  xb[i] = f2bf(emb[(size_t)tok[b] * EMBED + e]);
}

// --------------------------------------------------- bf16 WMMA GEMM, M = 64
// C[m][n] = sum_k A[m][k] * W[n][k] + bias[n]   (A: [64][K] bf16, W: [N][K] bf16)
// Block = 256 threads = 8 waves; wave w owns N-tile blockIdx*128 + 16*w with
// 4 accumulators covering M=64 -> every W element loaded from global exactly
// once. A panels double-buffered in padded LDS via global_load_async_to_lds
// (no staging VGPRs, ASYNCcnt overlap); sched_barrier groups 4 back-to-back
// WMMAs per K-step.  Requires N % 128 == 0, K % 128 == 0.
__global__ __launch_bounds__(256)
void gemm_bf16_wmma_kernel(const unsigned short* __restrict__ A, int lda,
                           const unsigned short* __restrict__ W, int ldb,
                           const float* __restrict__ bias,
                           float* __restrict__ C, long long ldc, int K) {
  __shared__ unsigned short As[2][64 * APITCH];

  const int tid  = threadIdx.x;
  const int wave = tid >> 5;
  const int lane = tid & 31;
  const int n0   = blockIdx.x * 128 + wave * 16;
  const int rsel = lane & 15;
  const int hi   = (lane >> 4) * 8;             // ISA 16-bit operand K-split

  // this thread's 4 staging chunks (8 bf16 each) of the 64 x KP panel
  int srow[4], sqc[4];
  unsigned ldst[2][4];
#pragma unroll
  for (int j = 0; j < 4; ++j) {
    int idx = tid + j * 256;
    srow[j] = idx >> 4;
    sqc[j]  = (idx & 15) * 8;
    ldst[0][j] = (unsigned)(uintptr_t)(&As[0][srow[j] * APITCH + sqc[j]]);
    ldst[1][j] = (unsigned)(uintptr_t)(&As[1][srow[j] * APITCH + sqc[j]]);
  }

  const unsigned short* pb = W + (size_t)(n0 + rsel) * ldb + hi;

  v8f acc0 = {}, acc1 = {}, acc2 = {}, acc3 = {};

  // prologue: stage panel 0 (async, direct to LDS)
#pragma unroll
  for (int j = 0; j < 4; ++j)
    async_ld_lds_b128(ldst[0][j], A + (size_t)srow[j] * lda + sqc[j]);
  wait_async0();
  __syncthreads();

  const int np = K / KP;
  for (int p = 0; p < np; ++p) {
    const int k0 = p * KP;

    if (p + 1 < np) {                           // async-fetch next A panel now;
#pragma unroll                                  // waited only at panel end
      for (int j = 0; j < 4; ++j)
        async_ld_lds_b128(ldst[(p + 1) & 1][j],
                          A + (size_t)srow[j] * lda + k0 + KP + sqc[j]);
      __builtin_prefetch(pb + k0 + KP, 0, 3);   // next B panel -> global_prefetch_b8
    }

    const unsigned short* Ap = &As[p & 1][0];
#pragma unroll
    for (int ks = 0; ks < KP; ks += 32) {
      union U { uint4 q[2]; v16bf v; } tb, ta0, ta1, ta2, ta3;
      tb.q[0] = *(const uint4*)(pb + k0 + ks);        // K = +hi .. +hi+7
      tb.q[1] = *(const uint4*)(pb + k0 + ks + 16);   // K = +16+hi .. +7
      const unsigned short* pr = Ap + rsel * APITCH + ks + hi;
      ta0.q[0] = *(const uint4*)(pr);
      ta0.q[1] = *(const uint4*)(pr + 16);
      ta1.q[0] = *(const uint4*)(pr + 16 * APITCH);
      ta1.q[1] = *(const uint4*)(pr + 16 * APITCH + 16);
      ta2.q[0] = *(const uint4*)(pr + 32 * APITCH);
      ta2.q[1] = *(const uint4*)(pr + 32 * APITCH + 16);
      ta3.q[0] = *(const uint4*)(pr + 48 * APITCH);
      ta3.q[1] = *(const uint4*)(pr + 48 * APITCH + 16);
      __builtin_amdgcn_sched_barrier(0);        // loads above, WMMAs below
      acc0 = WMMA_BF16(ta0.v, tb.v, acc0);      // 4 independent chains,
      acc1 = WMMA_BF16(ta1.v, tb.v, acc1);      // issue back-to-back
      acc2 = WMMA_BF16(ta2.v, tb.v, acc2);
      acc3 = WMMA_BF16(ta3.v, tb.v, acc3);
    }

    if (p + 1 < np) {                           // publish next panel
      wait_async0();
      __syncthreads();                          // single barrier per panel
    }
  }

  const int   col = n0 + rsel;
  const float bv  = bias[col];
  v8f accs[4] = { acc0, acc1, acc2, acc3 };
#pragma unroll
  for (int mt = 0; mt < 4; ++mt) {
    const int mrow = mt * 16 + hi;              // lanes>=16 hold M = 8+j
#pragma unroll
    for (int j = 0; j < 8; ++j)
      C[(size_t)(mrow + j) * ldc + col] = accs[mt][j] + bv;
  }
}

// ---------------------------------------------------------------- GRU cell
__global__ void gru_cell_kernel(const float* __restrict__ gi,
                                const float* __restrict__ gh,
                                float* __restrict__ h,
                                unsigned short* __restrict__ hb) {
  int i = blockIdx.x * blockDim.x + threadIdx.x; // NB*HIDDEN threads
  int b = i / HIDDEN, j = i % HIDDEN;
  const float* gib = gi + (size_t)b * G3;
  const float* ghb = gh + (size_t)b * G3;
  float r  = 1.0f / (1.0f + expf(-(gib[j] + ghb[j])));
  float z  = 1.0f / (1.0f + expf(-(gib[HIDDEN + j] + ghb[HIDDEN + j])));
  float n  = tanhf(gib[2 * HIDDEN + j] + r * ghb[2 * HIDDEN + j]);
  float hn = (1.0f - z) * n + z * h[i];
  h[i]  = hn;
  hb[i] = f2bf(hn);
}

// ------------------------------------------- log-softmax + greedy argmax
__global__ __launch_bounds__(256)
void logsoftmax_argmax_kernel(float* __restrict__ out, int t,
                              int* __restrict__ tok) {
  __shared__ float sm[256];
  __shared__ int   si[256];
  __shared__ float ss[256];
  const int b = blockIdx.x, tid = threadIdx.x;
  float* row = out + (size_t)b * (MAXLEN + 1) * VOCAB + (size_t)t * VOCAB;

  float m = -3.4e38f; int mi = 0;
  for (int v = tid; v < VOCAB; v += 256) {
    float x = row[v];
    if (x > m) { m = x; mi = v; }
  }
  sm[tid] = m; si[tid] = mi;
  __syncthreads();
  for (int s = 128; s > 0; s >>= 1) {
    if (tid < s) {
      if (sm[tid + s] > sm[tid] ||
          (sm[tid + s] == sm[tid] && si[tid + s] < si[tid])) {
        sm[tid] = sm[tid + s]; si[tid] = si[tid + s];
      }
    }
    __syncthreads();
  }
  const float mx = sm[0];

  float lsum = 0.0f;
  for (int v = tid; v < VOCAB; v += 256) lsum += expf(row[v] - mx);
  ss[tid] = lsum;
  __syncthreads();
  for (int s = 128; s > 0; s >>= 1) {
    if (tid < s) ss[tid] += ss[tid + s];
    __syncthreads();
  }
  const float lse = mx + logf(ss[0]);

  for (int v = tid; v < VOCAB; v += 256) row[v] = row[v] - lse;
  if (tid == 0) tok[b] = si[0];
}

// ------------------------------------------------------------- h_last copy
__global__ void copy_h_kernel(const float* __restrict__ h,
                              float* __restrict__ out) {
  int i = blockIdx.x * blockDim.x + threadIdx.x;
  if (i < NB * HIDDEN)
    out[(size_t)NB * (MAXLEN + 1) * VOCAB + i] = h[i];
}

// --------------------------------------------------------------- launcher
extern "C" void kernel_launch(void* const* d_in, const int* in_sizes, int n_in,
                              void* d_out, int out_size, void* d_ws, size_t ws_size,
                              hipStream_t stream) {
  // setup_inputs order: target, encoder_hidden, embedding, W_ih, W_hh,
  //                     b_ih, b_hh, fc_w, fc_b   (target unused by reference)
  const float* enc  = (const float*)d_in[1];
  const float* emb  = (const float*)d_in[2];
  const float* W_ih = (const float*)d_in[3];
  const float* W_hh = (const float*)d_in[4];
  const float* b_ih = (const float*)d_in[5];
  const float* b_hh = (const float*)d_in[6];
  const float* fc_w = (const float*)d_in[7];
  const float* fc_b = (const float*)d_in[8];
  float* out = (float*)d_out;

  // workspace layout (all segment sizes multiples of 256 B)
  char* ws = (char*)d_ws;
  size_t off = 0;
  unsigned short* fcw_b = (unsigned short*)(ws + off); off += (size_t)VOCAB * HIDDEN * 2;
  unsigned short* wih_b = (unsigned short*)(ws + off); off += (size_t)G3 * EMBED * 2;
  unsigned short* whh_b = (unsigned short*)(ws + off); off += (size_t)G3 * HIDDEN * 2;
  float*          h     = (float*)(ws + off);          off += (size_t)NB * HIDDEN * 4;
  unsigned short* hb    = (unsigned short*)(ws + off); off += (size_t)NB * HIDDEN * 2;
  unsigned short* xb    = (unsigned short*)(ws + off); off += (size_t)NB * EMBED * 2;
  float*          gi    = (float*)(ws + off);          off += (size_t)NB * G3 * 4;
  float*          gh    = (float*)(ws + off);          off += (size_t)NB * G3 * 4;
  int*            tok   = (int*)(ws + off);            off += 256;

  // per-launch (deterministic) bf16 weight conversion; fc_w stays L2-resident
  cvt_bf16_kernel<<<(VOCAB * HIDDEN + 255) / 256, 256, 0, stream>>>(fc_w, fcw_b, VOCAB * HIDDEN);
  cvt_bf16_kernel<<<(G3 * EMBED  + 255) / 256, 256, 0, stream>>>(W_ih, wih_b, G3 * EMBED);
  cvt_bf16_kernel<<<(G3 * HIDDEN + 255) / 256, 256, 0, stream>>>(W_hh, whh_b, G3 * HIDDEN);

  init_kernel<<<(NB * VOCAB + 255) / 256, 256, 0, stream>>>(enc, out, h, hb, tok);

  for (int t = 0; t < MAXLEN; ++t) {
    embed_kernel<<<(NB * EMBED) / 256, 256, 0, stream>>>(emb, tok, xb);

    // gi = x @ W_ih^T + b_ih   [64 x 3072], K = 512
    gemm_bf16_wmma_kernel<<<G3 / 128, 256, 0, stream>>>(
        xb, EMBED, wih_b, EMBED, b_ih, gi, (long long)G3, EMBED);
    // gh = h @ W_hh^T + b_hh   [64 x 3072], K = 1024
    gemm_bf16_wmma_kernel<<<G3 / 128, 256, 0, stream>>>(
        hb, HIDDEN, whh_b, HIDDEN, b_hh, gh, (long long)G3, HIDDEN);

    gru_cell_kernel<<<(NB * HIDDEN) / 256, 256, 0, stream>>>(gi, gh, h, hb);

    // logits -> out[b][t][:]   [64 x 32000], K = 1024
    gemm_bf16_wmma_kernel<<<VOCAB / 128, 256, 0, stream>>>(
        hb, HIDDEN, fcw_b, HIDDEN, fc_b,
        out + (size_t)t * VOCAB, (long long)(MAXLEN + 1) * VOCAB, HIDDEN);

    logsoftmax_argmax_kernel<<<NB, 256, 0, stream>>>(out, t, tok);
  }

  copy_h_kernel<<<(NB * HIDDEN + 255) / 256, 256, 0, stream>>>(h, out);
}
